// RWKVAttention_45337674777325
// MI455X (gfx1250) — matrix operand order repaired and checked by reading, not compiled, not run on hardware
//
#include <hip/hip_runtime.h>
#include <hip/hip_bf16.h>
#include <cstdint>

#define B_ 4
#define T_ 2048
#define C_ 2048
#define H_ 16
#define D_ 128
#define M_ (B_*T_)   // 8192 rows

typedef __attribute__((ext_vector_type(16))) __bf16 v16bf;
typedef __attribute__((ext_vector_type(8)))  float  v8f;

// Native bf16 conversions: let clang pick the gfx1250 cvt instructions.
__device__ __forceinline__ uint16_t f2bf(float x) {
    __bf16 h = (__bf16)x;
    return __builtin_bit_cast(uint16_t, h);
}
__device__ __forceinline__ float bf2f(uint16_t u) {
    __bf16 h = __builtin_bit_cast(__bf16, u);
    return (float)h;
}
__device__ __forceinline__ uint32_t pack2(float a, float b) {
    return (uint32_t)f2bf(a) | ((uint32_t)f2bf(b) << 16);
}
__device__ __forceinline__ void unpack8(uint4 u, float* f) {
    uint32_t w0 = u.x, w1 = u.y, w2 = u.z, w3 = u.w;
    f[0] = bf2f((uint16_t)(w0 & 0xFFFF)); f[1] = bf2f((uint16_t)(w0 >> 16));
    f[2] = bf2f((uint16_t)(w1 & 0xFFFF)); f[3] = bf2f((uint16_t)(w1 >> 16));
    f[4] = bf2f((uint16_t)(w2 & 0xFFFF)); f[5] = bf2f((uint16_t)(w2 >> 16));
    f[6] = bf2f((uint16_t)(w3 & 0xFFFF)); f[7] = bf2f((uint16_t)(w3 >> 16));
}

#define LDSTRIDE 72   // padded bf16 stride: conflict-free ds_read_b128 across 16 rows

// ---------------------------------------------------------------------------
// Stage one 128x64 A tile (token-shift mix fused, f32->bf16) and one 128x64
// W tile into LDS.
// ---------------------------------------------------------------------------
template<int AMODE>
__device__ __forceinline__ void stage_tiles(
    uint16_t* As, uint16_t* Bs, int tid, int blockM, int blockN, int k0,
    const float* __restrict__ X, const float* __restrict__ mix,
    const uint16_t* __restrict__ Abf, const float* __restrict__ W)
{
    #pragma unroll
    for (int it = 0; it < 8; ++it) {
        int idx = tid + it * 256;           // 0..2047 float4 chunks
        int row = idx >> 4;                 // 0..127
        int c4  = (idx & 15) << 2;          // 0,4,...,60
        int mg  = blockM + row;

        uint2 pa;
        if (AMODE == 0) {
            int t    = mg & (T_ - 1);
            int prow = (t == 0) ? (mg + 1) : (mg - 1);
            float4 cur = *(const float4*)(X + (size_t)mg   * C_ + k0 + c4);
            float4 prv = *(const float4*)(X + (size_t)prow * C_ + k0 + c4);
            float4 mx  = *(const float4*)(mix + k0 + c4);
            pa.x = pack2(cur.x + (prv.x - cur.x) * mx.x,
                         cur.y + (prv.y - cur.y) * mx.y);
            pa.y = pack2(cur.z + (prv.z - cur.z) * mx.z,
                         cur.w + (prv.w - cur.w) * mx.w);
        } else {
            pa = *(const uint2*)(Abf + (size_t)mg * C_ + k0 + c4);
        }
        *(uint2*)(As + row * LDSTRIDE + c4) = pa;

        float4 wv = *(const float4*)(W + (size_t)(blockN + row) * C_ + k0 + c4);
        uint2 pw;
        pw.x = pack2(wv.x, wv.y);
        pw.y = pack2(wv.z, wv.w);
        *(uint2*)(Bs + row * LDSTRIDE + c4) = pw;
    }
}

// ---------------------------------------------------------------------------
// Tiled bf16 WMMA GEMM:  out[m,n] = epilogue( sum_k A[m,k] * W[n,k] )
// AMODE 0: A built on the fly from x + (tokenshift(x)-x)*mix  (f32 -> bf16)
// AMODE 1: A read directly as bf16 (the gated attention buffer)
// EP 0: store bf16 | EP 1: sigmoid, store bf16 | EP 2: store f32
// Block tile 128x128, 8 waves (wave32), wave tile 32x64 = 2x4 WMMA 16x16x32.
// LDS double-buffered: one barrier per 64-deep K block.
// ---------------------------------------------------------------------------
template<int AMODE, int EP>
__global__ __launch_bounds__(256) void wmma_gemm_kernel(
    const float* __restrict__ X, const float* __restrict__ mix,
    const uint16_t* __restrict__ Abf, const float* __restrict__ W,
    void* __restrict__ out)
{
    __shared__ __align__(16) uint16_t As[2][128 * LDSTRIDE];
    __shared__ __align__(16) uint16_t Bs[2][128 * LDSTRIDE];

    const int tid  = threadIdx.x;
    const int lane = tid & 31, wave = tid >> 5;
    const int l16  = lane & 15, lh = lane >> 4;
    const int wm   = wave & 3,  wn = wave >> 2;
    const int blockN = blockIdx.x * 128;
    const int blockM = blockIdx.y * 128;

    v8f acc[2][4];
    #pragma unroll
    for (int i = 0; i < 2; ++i)
        #pragma unroll
        for (int j = 0; j < 4; ++j)
            #pragma unroll
            for (int q = 0; q < 8; ++q) acc[i][j][q] = 0.0f;

    stage_tiles<AMODE>(As[0], Bs[0], tid, blockM, blockN, 0, X, mix, Abf, W);
    __syncthreads();

    const int NKB = C_ / 64;   // 32
    for (int kb = 0; kb < NKB; ++kb) {
        const int cur = kb & 1;
        if (kb + 1 < NKB)
            stage_tiles<AMODE>(As[cur ^ 1], Bs[cur ^ 1], tid, blockM, blockN,
                               (kb + 1) * 64, X, mix, Abf, W);

        const uint16_t* Ab = As[cur];
        const uint16_t* Bb = Bs[cur];
        #pragma unroll
        for (int kk = 0; kk < 2; ++kk) {
            union FragU { uint4 u[2]; v16bf v; };
            FragU a[2], b[4];
            #pragma unroll
            for (int i = 0; i < 2; ++i) {
                int mrow = wm * 32 + i * 16 + l16;
                // A frag (16x32): v0..3 = 8 bf16 @ k=lh*8, v4..7 @ k=16+lh*8
                a[i].u[0] = *(const uint4*)(Ab + mrow * LDSTRIDE + kk * 32 + lh * 8);
                a[i].u[1] = *(const uint4*)(Ab + mrow * LDSTRIDE + kk * 32 + 16 + lh * 8);
            }
            #pragma unroll
            for (int j = 0; j < 4; ++j) {
                int nrow = wn * 64 + j * 16 + l16;
                // B frag (32x16): lane n=l16, v0..3 = 8 bf16 @ k=lh*8, v4..7 @ k=16+lh*8
                b[j].u[0] = *(const uint4*)(Bb + nrow * LDSTRIDE + kk * 32 + lh * 8);
                b[j].u[1] = *(const uint4*)(Bb + nrow * LDSTRIDE + kk * 32 + 16 + lh * 8);
            }
            #pragma unroll
            for (int i = 0; i < 2; ++i)
                #pragma unroll
                for (int j = 0; j < 4; ++j)
                    acc[i][j] = __builtin_amdgcn_wmma_f32_16x16x32_bf16(
                        false, a[i].v, false, b[j].v, (short)0, acc[i][j],
                        false, false);
        }
        __syncthreads();
    }

    // ---- epilogue: D layout -> VGPR q holds row q + lh*8, col l16 ----
    float*    outF = (float*)out;
    uint16_t* outH = (uint16_t*)out;
    #pragma unroll
    for (int i = 0; i < 2; ++i) {
        #pragma unroll
        for (int j = 0; j < 4; ++j) {
            int n = blockN + wn * 64 + j * 16 + l16;
            #pragma unroll
            for (int q = 0; q < 8; ++q) {
                int m = blockM + wm * 32 + i * 16 + lh * 8 + q;
                float val = acc[i][j][q];
                if (EP == 1) val = 1.0f / (1.0f + __expf(-val));
                if (EP == 2) outF[(size_t)m * C_ + n] = val;
                else         outH[(size_t)m * C_ + n] = f2bf(val);
            }
        }
    }
}

// ---------------------------------------------------------------------------
// RWKV scan: one block per (b,h). 256 threads tile the 128x128 state into
// 8x8 register blocks. Per step: state = state*w + k⊗v; out = state·r.
// Output written directly into the reference's (B,H,T,D)-as-(B,T,C) view,
// fused with the sigmoid gate g -> bf16 "ag" buffer.
// ---------------------------------------------------------------------------
__global__ __launch_bounds__(256) void rwkv_scan_kernel(
    const uint16_t* __restrict__ r, const uint16_t* __restrict__ k,
    const uint16_t* __restrict__ v, const uint16_t* __restrict__ g,
    const float* __restrict__ time_decay, uint16_t* __restrict__ ag)
{
    const int bh = blockIdx.x;
    const int b = bh >> 4, h = bh & 15;
    const int tid = threadIdx.x;
    const int tj = tid & 15, ti = tid >> 4;   // rows ti*8..+8, cols tj*8..+8

    float w8[8];
    #pragma unroll
    for (int q = 0; q < 8; ++q)
        w8[q] = __expf(-__expf(time_decay[h * D_ + ti * 8 + q]));

    float s[8][8];
    #pragma unroll
    for (int q = 0; q < 8; ++q)
        #pragma unroll
        for (int p = 0; p < 8; ++p) s[q][p] = 0.0f;

    const size_t rkv_base = (size_t)(b * T_) * C_ + h * D_;
    const uint16_t* kp = k + rkv_base + ti * 8;
    const uint16_t* vp = v + rkv_base + tj * 8;
    const uint16_t* rp = r + rkv_base + tj * 8;

    uint4 kq = *(const uint4*)kp;
    uint4 vq = *(const uint4*)vp;
    uint4 rq = *(const uint4*)rp;

    for (int t = 0; t < T_; ++t) {
        // deep prefetch: pull cachelines ~8 steps ahead (global_prefetch_b8)
        if ((t & 7) == 0 && t + 8 < T_) {
            size_t po = (size_t)(t + 8) * C_;
            __builtin_prefetch(kp + po, 0, 0);
            __builtin_prefetch(vp + po, 0, 0);
            __builtin_prefetch(rp + po, 0, 0);
        }
        uint4 kn = kq, vn = vq, rn = rq;
        if (t + 1 < T_) {                       // software-pipelined next-step load
            size_t o = (size_t)(t + 1) * C_;
            kn = *(const uint4*)(kp + o);
            vn = *(const uint4*)(vp + o);
            rn = *(const uint4*)(rp + o);
        }
        float kf[8], vf[8], rf[8];
        unpack8(kq, kf); unpack8(vq, vf); unpack8(rq, rf);

        float psum[8];
        #pragma unroll
        for (int q = 0; q < 8; ++q) {
            float p = 0.0f;
            #pragma unroll
            for (int pc = 0; pc < 8; ++pc) {
                s[q][pc] = s[q][pc] * w8[q] + kf[q] * vf[pc];
                p += s[q][pc] * rf[pc];
            }
            psum[q] = p;
        }
        // reduce over the 16 column-group lanes (wave32 halves)
        #pragma unroll
        for (int m = 1; m < 16; m <<= 1)
            #pragma unroll
            for (int q = 0; q < 8; ++q)
                psum[q] += __shfl_xor(psum[q], m, 32);

        if (tj < 8) {
            float o = psum[0];
            if (tj == 1) o = psum[1];
            if (tj == 2) o = psum[2];
            if (tj == 3) o = psum[3];
            if (tj == 4) o = psum[4];
            if (tj == 5) o = psum[5];
            if (tj == 6) o = psum[6];
            if (tj == 7) o = psum[7];
            int i = ti * 8 + tj;
            // scrambled view: attn[b, (h*T+t)*D + i] within batch b
            size_t flat = (size_t)b * T_ * C_ + ((size_t)h * T_ + t) * D_ + i;
            ag[flat] = f2bf(o * bf2f(g[flat]));
        }
        kq = kn; vq = vn; rq = rn;
    }
}

// ---------------------------------------------------------------------------
// LayerNorm over last dim (C=2048): one block per row, 256 threads x 8 elems.
// ---------------------------------------------------------------------------
__global__ __launch_bounds__(256) void layernorm_kernel(
    const float* __restrict__ y, const float* __restrict__ lnw,
    const float* __restrict__ lnb, float* __restrict__ out)
{
    __shared__ float ssum[256];
    __shared__ float ssq[256];
    const int row = blockIdx.x;
    const int tid = threadIdx.x;
    const float* yr = y + (size_t)row * C_;

    float vv[8];
    #pragma unroll
    for (int i = 0; i < 2; ++i) {
        float4 t4 = *(const float4*)(yr + tid * 8 + i * 4);
        vv[i*4+0] = t4.x; vv[i*4+1] = t4.y; vv[i*4+2] = t4.z; vv[i*4+3] = t4.w;
    }
    float s = 0.0f, q2 = 0.0f;
    #pragma unroll
    for (int i = 0; i < 8; ++i) { s += vv[i]; q2 += vv[i] * vv[i]; }
    ssum[tid] = s; ssq[tid] = q2;
    __syncthreads();
    for (int off = 128; off > 0; off >>= 1) {
        if (tid < off) { ssum[tid] += ssum[tid + off]; ssq[tid] += ssq[tid + off]; }
        __syncthreads();
    }
    float mu   = ssum[0] * (1.0f / C_);
    float var  = ssq[0] * (1.0f / C_) - mu * mu;
    float rstd = rsqrtf(var + 1e-5f);
    #pragma unroll
    for (int i = 0; i < 8; ++i) {
        int c = tid * 8 + i;
        out[(size_t)row * C_ + c] = (vv[i] - mu) * rstd * lnw[c] + lnb[c];
    }
}

// ---------------------------------------------------------------------------
extern "C" void kernel_launch(void* const* d_in, const int* in_sizes, int n_in,
                              void* d_out, int out_size, void* d_ws, size_t ws_size,
                              hipStream_t stream) {
    (void)in_sizes; (void)n_in; (void)out_size; (void)ws_size;
    const float* x   = (const float*)d_in[0];
    const float* td  = (const float*)d_in[1];
    const float* mk  = (const float*)d_in[2];
    const float* mv  = (const float*)d_in[3];
    const float* mr  = (const float*)d_in[4];
    const float* mgx = (const float*)d_in[5];
    const float* Wr  = (const float*)d_in[6];
    const float* Wk  = (const float*)d_in[7];
    const float* Wv  = (const float*)d_in[8];
    const float* Wg  = (const float*)d_in[9];
    const float* Wo  = (const float*)d_in[10];
    const float* lnw = (const float*)d_in[11];
    const float* lnb = (const float*)d_in[12];

    const size_t NE  = (size_t)M_ * C_;            // 16,777,216 elements
    const size_t SZH = NE * sizeof(uint16_t);      // 32 MiB per bf16 buffer
    char* ws = (char*)d_ws;
    uint16_t* rb  = (uint16_t*)(ws + 0 * SZH);
    uint16_t* kb  = (uint16_t*)(ws + 1 * SZH);
    uint16_t* vb  = (uint16_t*)(ws + 2 * SZH);
    uint16_t* gb  = (uint16_t*)(ws + 3 * SZH);
    uint16_t* agb = (uint16_t*)(ws + 4 * SZH);
    float*    yb  = (float*)   (ws + 5 * SZH);     // 64 MiB f32

    dim3 gg(C_ / 128, M_ / 128);                   // 16 x 64 blocks
    wmma_gemm_kernel<0, 0><<<gg, 256, 0, stream>>>(x, mr,  nullptr, Wr, rb);
    wmma_gemm_kernel<0, 0><<<gg, 256, 0, stream>>>(x, mk,  nullptr, Wk, kb);
    wmma_gemm_kernel<0, 0><<<gg, 256, 0, stream>>>(x, mv,  nullptr, Wv, vb);
    wmma_gemm_kernel<0, 1><<<gg, 256, 0, stream>>>(x, mgx, nullptr, Wg, gb);
    rwkv_scan_kernel<<<B_ * H_, 256, 0, stream>>>(rb, kb, vb, gb, td, agb);
    wmma_gemm_kernel<1, 2><<<gg, 256, 0, stream>>>(nullptr, nullptr, agb, Wo, yb);
    layernorm_kernel<<<M_, 256, 0, stream>>>(yb, lnw, lnb, (float*)d_out);
}
